// Model_77747497992554
// MI455X (gfx1250) — matrix operand order repaired and checked by reading, not compiled, not run on hardware
//
#include <hip/hip_runtime.h>
#include <hip/hip_bf16.h>
#include <stdint.h>

// ---------------------------------------------------------------------------
// TabR-style retrieval model for MI455X (gfx1250, wave32, WMMA).
// All GEMMs run through v_wmma_f32_16x16x32_f16 with f32 accumulation.
// Epilogue options are compile-time template flags so stores are branch-free.
// ---------------------------------------------------------------------------

typedef __attribute__((ext_vector_type(16))) _Float16 v16h;
typedef __attribute__((ext_vector_type(8)))  _Float16 v8h;
typedef __attribute__((ext_vector_type(8)))  float    v8f;

#define D_MAIN 256
#define D_HID  512
#define N_NUM  96
#define NBATCH 1024
#define N_CAND 100000
#define CTX    96
#define LDA    520   // f16 LDS row stride (multiple of 8 -> 16B aligned rows)

// ---------------------------------------------------------------------------
// Block GEMM: OUT[16 x N] = IN[16 x K] @ W^T  (+bias, +residual, optional ReLU)
// IN is f16 in LDS (row stride ldIn), W^T is f16 global [N][K] row-major.
// 8 waves, each wave owns N/(16*8) column tiles. Caller provides barriers.
// A fragment (16x32 f16): lane l: row=l&15, K = (e>>3)*16 + (l>>4)*8 + (e&7).
// B fragment (32x16 f16): lane l: col=l&15, K = (e>>3)*16 + (l>>4)*8 + (e&7).
// C/D (16x16 f32): lane l: col=l&15, row = (l>>4)*8 + r.
// ---------------------------------------------------------------------------
template<int K, int N, bool BIAS, bool RELU, bool O32, bool O16, bool RESID>
__device__ inline void gemm_block(const _Float16* __restrict__ sIn, int ldIn,
                                  const _Float16* __restrict__ wt,
                                  const float* __restrict__ bias,
                                  float* __restrict__ outF32, int ldO32,
                                  _Float16* __restrict__ outF16, int ldO16,
                                  const float* __restrict__ resid, int ldR)
{
    const int tid  = threadIdx.x;
    const int wave = tid >> 5;
    const int lane = tid & 31;
    const int col  = lane & 15;
    const int hi   = lane >> 4;
    constexpr int NT = N / (16 * 8);

    v8f acc[NT];
    v8f zero = {};
#pragma unroll
    for (int nt = 0; nt < NT; ++nt) acc[nt] = zero;

    for (int kk = 0; kk < K; kk += 32) {
        // A fragment from LDS (two 16B loads)
        const _Float16* pa = sIn + col * ldIn + kk + hi * 8;
        v8h a0 = *(const v8h*)pa;
        v8h a1 = *(const v8h*)(pa + 16);
        v16h a;
#pragma unroll
        for (int i = 0; i < 8; ++i) { a[i] = a0[i]; a[8 + i] = a1[i]; }

#pragma unroll
        for (int nt = 0; nt < NT; ++nt) {
            int n0 = (wave * NT + nt) * 16;
            const _Float16* pb = wt + (size_t)(n0 + col) * K + kk + hi * 8;
            if (nt == 0 && kk + 32 < K)
                __builtin_prefetch(pb + 32, 0, 1);   // global_prefetch_b8
            v8h b0 = *(const v8h*)pb;
            v8h b1 = *(const v8h*)(pb + 16);
            v16h b;
#pragma unroll
            for (int i = 0; i < 8; ++i) { b[i] = b0[i]; b[8 + i] = b1[i]; }
            acc[nt] = __builtin_amdgcn_wmma_f32_16x16x32_f16(
                false, a, false, b, (short)0, acc[nt], false, false);
        }
    }

#pragma unroll
    for (int nt = 0; nt < NT; ++nt) {
        int n0 = (wave * NT + nt) * 16;
        int c  = n0 + col;
        float bv = BIAS ? bias[c] : 0.0f;
#pragma unroll
        for (int r = 0; r < 8; ++r) {
            int row = hi * 8 + r;
            float v = acc[nt][r] + bv;
            if (RESID) v += resid[row * ldR + c];
            if (RELU)  v  = fmaxf(v, 0.0f);
            if (O32)   outF32[row * ldO32 + c] = v;
            if (O16)   outF16[row * ldO16 + c] = (_Float16)v;
        }
    }
}

// ---------------------------------------------------------------------------
// Weight prep: W[K][N] f32 -> W^T[N][K] f16
// ---------------------------------------------------------------------------
__global__ void transpose_f16_kernel(const float* __restrict__ W,
                                     _Float16* __restrict__ Wt,
                                     int K, int N)
{
    int i = blockIdx.x * blockDim.x + threadIdx.x;
    if (i < K * N) {
        int k = i / N, n = i % N;
        Wt[(size_t)n * K + k] = (_Float16)W[i];
    }
}

// ---------------------------------------------------------------------------
// Fused encoder: X[rows x 96] -> h (residual MLP) -> LN -> keys.
// One block = 16 rows. QUERY selects the extra f32 outputs vs. ||k||^2.
// ---------------------------------------------------------------------------
template<bool QUERY>
__global__ __launch_bounds__(256) void encode_kernel(
    const float* __restrict__ X,
    const _Float16* __restrict__ wt_in, const float* __restrict__ b_in,
    const _Float16* __restrict__ wt_e1, const float* __restrict__ be1,
    const _Float16* __restrict__ wt_e2, const float* __restrict__ be2,
    const float* __restrict__ g_m, const float* __restrict__ b_m,
    const _Float16* __restrict__ wt_s, const float* __restrict__ bs,
    float* __restrict__ out_h,        // [rows x 256]   (QUERY)
    float* __restrict__ out_kf,       // [rows x 256]   (QUERY)
    _Float16* __restrict__ out_kh,    // [rows x 256] f16 keys
    float* __restrict__ out_n2)       // [rows] ||k||^2 (!QUERY)
{
    __shared__ __attribute__((aligned(16))) _Float16 sA[16 * LDA];
    __shared__ __attribute__((aligned(16))) _Float16 sB[16 * LDA];
    __shared__ float sH[16 * 256];
    __shared__ float sRed[256], sRed2[256], sStat[32];

    const int tid = threadIdx.x;
    const int row = tid >> 4, seg = tid & 15;
    const size_t rb = (size_t)blockIdx.x * 16;

    // stage X tile (16 x 96) as f16 into sA
#pragma unroll
    for (int i = 0; i < 6; ++i) {
        int c = seg * 6 + i;
        sA[row * LDA + c] = (_Float16)X[(rb + row) * N_NUM + c];
    }
    __syncthreads();

    // h = X @ W_in + b_in      (f32 -> sH, f16 -> sB)
    gemm_block<N_NUM, D_MAIN, true, false, true, true, false>(
        sA, LDA, wt_in, b_in, sH, 256, sB, LDA, nullptr, 0);
    __syncthreads();
    // t = ReLU(h @ We1 + be1)  (f16 -> sA)
    gemm_block<D_MAIN, D_HID, true, true, false, true, false>(
        sB, LDA, wt_e1, be1, nullptr, 0, sA, LDA, nullptr, 0);
    __syncthreads();
    // h2 = h + t @ We2 + be2   (f32 -> sH, residual sH)
    gemm_block<D_HID, D_MAIN, true, false, true, false, true>(
        sA, LDA, wt_e2, be2, sH, 256, nullptr, 0, sH, 256);
    __syncthreads();

    if (QUERY) {
#pragma unroll
        for (int i = 0; i < 16; ++i) {
            int c = seg * 16 + i;
            out_h[(rb + row) * 256 + c] = sH[row * 256 + c];
        }
    }

    // LayerNorm(h2) * g_m + b_m  -> f16 sB
    {
        float s = 0.f, s2 = 0.f;
#pragma unroll
        for (int i = 0; i < 16; ++i) {
            float v = sH[row * 256 + seg * 16 + i];
            s += v; s2 += v * v;
        }
        sRed[row * 16 + seg] = s; sRed2[row * 16 + seg] = s2;
        __syncthreads();
        if (tid < 16) {
            float a = 0.f, b = 0.f;
            for (int i = 0; i < 16; ++i) { a += sRed[tid * 16 + i]; b += sRed2[tid * 16 + i]; }
            float mu  = a * (1.0f / 256.0f);
            float var = b * (1.0f / 256.0f) - mu * mu;
            sStat[tid * 2]     = mu;
            sStat[tid * 2 + 1] = rsqrtf(var + 1e-5f);
        }
        __syncthreads();
        float mu = sStat[row * 2], rstd = sStat[row * 2 + 1];
#pragma unroll
        for (int i = 0; i < 16; ++i) {
            int c = seg * 16 + i;
            float v = (sH[row * 256 + c] - mu) * rstd * g_m[c] + b_m[c];
            sB[row * LDA + c] = (_Float16)v;
        }
    }
    __syncthreads();

    // k = ln @ Ws + bs  (f32 -> sH)
    gemm_block<D_MAIN, D_MAIN, true, false, true, false, false>(
        sB, LDA, wt_s, bs, sH, 256, nullptr, 0, nullptr, 0);
    __syncthreads();

    // epilogue: emit keys (+ squared norms for candidates)
    {
        float s2 = 0.f;
#pragma unroll
        for (int i = 0; i < 16; ++i) {
            int c = seg * 16 + i;
            float v = sH[row * 256 + c];
            if (QUERY) out_kf[(rb + row) * 256 + c] = v;
            out_kh[(rb + row) * 256 + c] = (_Float16)v;
            s2 += v * v;
        }
        if (!QUERY) {
            sRed[row * 16 + seg] = s2;
            __syncthreads();
            if (tid < 16) {
                float a = 0.f;
                for (int i = 0; i < 16; ++i) a += sRed[tid * 16 + i];
                out_n2[rb + tid] = a;
            }
        }
    }
}

// ---------------------------------------------------------------------------
// Streaming similarity + top-96 + softmax. One block = 16 query rows.
// score = 2*k.ck - ||ck||^2   (the -||k||^2 term cancels in softmax).
// 8 waves each WMMA one 16x16 candidate tile per iteration; 16 threads
// maintain per-row top-96 lists in LDS.
// ---------------------------------------------------------------------------
__global__ __launch_bounds__(256) void topk_kernel(
    const _Float16* __restrict__ qk_h, const _Float16* __restrict__ ck_h,
    const float* __restrict__ ck_n2,
    float* __restrict__ probs, int* __restrict__ topidx)
{
    __shared__ float sS[8 * 256];
    __shared__ float sVal[16 * CTX];
    __shared__ int   sIdx[16 * CTX];

    const int tid  = threadIdx.x;
    const int wave = tid >> 5, lane = tid & 31;
    const int col  = lane & 15, hi = lane >> 4;
    const size_t qb = (size_t)blockIdx.x * 16;

    // keep the 8 query A-fragments (K=256) in registers for the whole sweep
    v16h afrag[8];
#pragma unroll
    for (int k8 = 0; k8 < 8; ++k8) {
        const _Float16* pa = qk_h + (qb + col) * 256 + k8 * 32 + hi * 8;
        v8h a0 = *(const v8h*)pa;
        v8h a1 = *(const v8h*)(pa + 16);
#pragma unroll
        for (int i = 0; i < 8; ++i) { afrag[k8][i] = a0[i]; afrag[k8][8 + i] = a1[i]; }
    }

    float hvmin = 3.0e38f; int hpmin = 0, hcnt = 0;
    const int NTILE = N_CAND / 16;                 // 6250
    const int NITER = (NTILE + 7) / 8;             // 782

    for (int iter = 0; iter < NITER; ++iter) {
        int tile = iter * 8 + wave;
        if (tile < NTILE) {
            int c0 = tile * 16;
            v8f acc = {};
#pragma unroll
            for (int k8 = 0; k8 < 8; ++k8) {
                const _Float16* pb = ck_h + (size_t)(c0 + col) * 256 + k8 * 32 + hi * 8;
                v8h b0 = *(const v8h*)pb;
                v8h b1 = *(const v8h*)(pb + 16);
                v16h b;
#pragma unroll
                for (int i = 0; i < 8; ++i) { b[i] = b0[i]; b[8 + i] = b1[i]; }
                acc = __builtin_amdgcn_wmma_f32_16x16x32_f16(
                    false, afrag[k8], false, b, (short)0, acc, false, false);
            }
            float n2 = ck_n2[c0 + col];
#pragma unroll
            for (int r = 0; r < 8; ++r)
                sS[wave * 256 + (hi * 8 + r) * 16 + col] = 2.0f * acc[r] - n2;
        }
        __syncthreads();
        if (tid < 16) {
            int wmax = NTILE - iter * 8; if (wmax > 8) wmax = 8;
            for (int w = 0; w < wmax; ++w) {
                int cbase = (iter * 8 + w) * 16;
                for (int c = 0; c < 16; ++c) {
                    float s = sS[w * 256 + tid * 16 + c];
                    if (hcnt < CTX) {
                        sVal[tid * CTX + hcnt] = s;
                        sIdx[tid * CTX + hcnt] = cbase + c;
                        if (s < hvmin) { hvmin = s; hpmin = hcnt; }
                        ++hcnt;
                    } else if (s > hvmin) {
                        sVal[tid * CTX + hpmin] = s;
                        sIdx[tid * CTX + hpmin] = cbase + c;
                        hvmin = sVal[tid * CTX]; hpmin = 0;
                        for (int i = 1; i < CTX; ++i)
                            if (sVal[tid * CTX + i] < hvmin) { hvmin = sVal[tid * CTX + i]; hpmin = i; }
                    }
                }
            }
        }
        __syncthreads();
    }

    // softmax over the (unordered) top-96 — permutation invariant
    if (tid < 16) {
        float mx = -3.0e38f;
        for (int i = 0; i < CTX; ++i) mx = fmaxf(mx, sVal[tid * CTX + i]);
        float sum = 0.f;
        for (int i = 0; i < CTX; ++i) {
            float e = __expf(sVal[tid * CTX + i] - mx);
            sVal[tid * CTX + i] = e; sum += e;
        }
        float inv = 1.0f / sum;
        for (int i = 0; i < CTX; ++i) {
            probs [(qb + tid) * CTX + i] = sVal[tid * CTX + i] * inv;
            topidx[(qb + tid) * CTX + i] = sIdx[tid * CTX + i];
        }
    }
}

// ---------------------------------------------------------------------------
// Value MLP + probability-weighted mix. One block = one query, loops over
// 6 tiles of 16 neighbors; no atomics.
// ---------------------------------------------------------------------------
__global__ __launch_bounds__(256) void values_kernel(
    const float* __restrict__ qk_f32, const _Float16* __restrict__ ck_h,
    const float* __restrict__ cand_y, const int* __restrict__ topidx,
    const float* __restrict__ probs,
    const _Float16* __restrict__ wt_t1,
    const _Float16* __restrict__ wt_t2, const float* __restrict__ bt2,
    const float* __restrict__ Wy, const float* __restrict__ by,
    const float* __restrict__ x_enc, float* __restrict__ x_mix)
{
    __shared__ __attribute__((aligned(16))) _Float16 sA[16 * LDA];
    __shared__ __attribute__((aligned(16))) _Float16 sB[16 * LDA];
    __shared__ float sH[16 * 256];

    const int tid = threadIdx.x;
    const int row = tid >> 4, seg = tid & 15;
    const int b = blockIdx.x;

    float acc = 0.f;   // this thread owns output column `tid`

    for (int mt = 0; mt < CTX / 16; ++mt) {
        int ci = topidx[b * CTX + mt * 16 + row];
#pragma unroll
        for (int i = 0; i < 16; ++i) {
            int c = seg * 16 + i;
            float d = qk_f32[b * 256 + c] - (float)ck_h[(size_t)ci * 256 + c];
            sB[row * LDA + c] = (_Float16)d;
        }
        __syncthreads();
        // t = ReLU(diff @ Wt1)          (no bias)
        gemm_block<D_MAIN, D_HID, false, true, false, true, false>(
            sB, LDA, wt_t1, nullptr, nullptr, 0, sA, LDA, nullptr, 0);
        __syncthreads();
        // v = t @ Wt2 + bt2
        gemm_block<D_HID, D_MAIN, true, false, true, false, false>(
            sA, LDA, wt_t2, bt2, sH, 256, nullptr, 0, nullptr, 0);
        __syncthreads();
        {
            int c = tid;
            float wyc = Wy[c], byc = by[c];
            for (int j = 0; j < 16; ++j) {
                int m = mt * 16 + j;
                int cj = topidx[b * CTX + m];
                float p = probs[b * CTX + m];
                acc += p * (sH[j * 256 + c] + wyc * cand_y[cj] + byc);
            }
        }
        __syncthreads();
    }
    x_mix[b * 256 + tid] = x_enc[b * 256 + tid] + acc;
}

// ---------------------------------------------------------------------------
// Predictor block + LN + ReLU + linear head. One block = 16 rows.
// ---------------------------------------------------------------------------
__global__ __launch_bounds__(256) void head_kernel(
    const float* __restrict__ x_mix,
    const _Float16* __restrict__ wt_p1, const float* __restrict__ bp1,
    const _Float16* __restrict__ wt_p2, const float* __restrict__ bp2,
    const float* __restrict__ g_h, const float* __restrict__ b_h,
    const float* __restrict__ Wh, const float* __restrict__ bh,
    float* __restrict__ out)
{
    __shared__ __attribute__((aligned(16))) _Float16 sA[16 * LDA];
    __shared__ __attribute__((aligned(16))) _Float16 sB[16 * LDA];
    __shared__ float sH[16 * 256];
    __shared__ float sRed[256], sRed2[256], sStat[32];

    const int tid = threadIdx.x;
    const int row = tid >> 4, seg = tid & 15;
    const size_t rb = (size_t)blockIdx.x * 16;

#pragma unroll
    for (int i = 0; i < 16; ++i) {
        int c = seg * 16 + i;
        float v = x_mix[(rb + row) * 256 + c];
        sH[row * 256 + c] = v;
        sB[row * LDA + c] = (_Float16)v;
    }
    __syncthreads();
    gemm_block<D_MAIN, D_HID, true, true, false, true, false>(
        sB, LDA, wt_p1, bp1, nullptr, 0, sA, LDA, nullptr, 0);
    __syncthreads();
    gemm_block<D_HID, D_MAIN, true, false, true, false, true>(
        sA, LDA, wt_p2, bp2, sH, 256, nullptr, 0, sH, 256);
    __syncthreads();

    // LN stats
    float s = 0.f, s2 = 0.f;
#pragma unroll
    for (int i = 0; i < 16; ++i) {
        float v = sH[row * 256 + seg * 16 + i];
        s += v; s2 += v * v;
    }
    sRed[row * 16 + seg] = s; sRed2[row * 16 + seg] = s2;
    __syncthreads();
    if (tid < 16) {
        float a = 0.f, bsum = 0.f;
        for (int i = 0; i < 16; ++i) { a += sRed[tid * 16 + i]; bsum += sRed2[tid * 16 + i]; }
        float mu  = a * (1.0f / 256.0f);
        float var = bsum * (1.0f / 256.0f) - mu * mu;
        sStat[tid * 2]     = mu;
        sStat[tid * 2 + 1] = rsqrtf(var + 1e-5f);
    }
    __syncthreads();
    float mu = sStat[row * 2], rstd = sStat[row * 2 + 1];
    float dot = 0.f;
#pragma unroll
    for (int i = 0; i < 16; ++i) {
        int c = seg * 16 + i;
        float v = (sH[row * 256 + c] - mu) * rstd * g_h[c] + b_h[c];
        v = fmaxf(v, 0.0f);
        dot += v * Wh[c];
    }
    __syncthreads();
    sRed[row * 16 + seg] = dot;
    __syncthreads();
    if (tid < 16) {
        float a = 0.f;
        for (int i = 0; i < 16; ++i) a += sRed[tid * 16 + i];
        out[rb + tid] = a + bh[0];
    }
}

// ---------------------------------------------------------------------------
// Host side
// ---------------------------------------------------------------------------
extern "C" void kernel_launch(void* const* d_in, const int* in_sizes, int n_in,
                              void* d_out, int out_size, void* d_ws, size_t ws_size,
                              hipStream_t stream)
{
    const float* x_num  = (const float*)d_in[0];
    const float* cand_x = (const float*)d_in[1];
    const float* cand_y = (const float*)d_in[2];
    // d_in[3] = context_size (96, hard-coded as CTX)
    const float* W_in = (const float*)d_in[4];
    const float* b_in = (const float*)d_in[5];
    const float* We1  = (const float*)d_in[6];
    const float* be1  = (const float*)d_in[7];
    const float* We2  = (const float*)d_in[8];
    const float* be2  = (const float*)d_in[9];
    const float* g_m  = (const float*)d_in[10];
    const float* b_m  = (const float*)d_in[11];
    const float* Ws   = (const float*)d_in[12];
    const float* bs   = (const float*)d_in[13];
    const float* Wy   = (const float*)d_in[14];
    const float* by   = (const float*)d_in[15];
    const float* Wt1  = (const float*)d_in[16];
    const float* Wt2  = (const float*)d_in[17];
    const float* bt2  = (const float*)d_in[18];
    const float* Wp1  = (const float*)d_in[19];
    const float* bp1  = (const float*)d_in[20];
    const float* Wp2  = (const float*)d_in[21];
    const float* bp2  = (const float*)d_in[22];
    const float* g_h  = (const float*)d_in[23];
    const float* b_h  = (const float*)d_in[24];
    const float* Wh   = (const float*)d_in[25];
    const float* bh   = (const float*)d_in[26];

    uint8_t* ws = (uint8_t*)d_ws;
    size_t off = 0;
    auto take = [&](size_t bytes) -> void* {
        void* p = ws + off;
        off += (bytes + 255) & ~(size_t)255;
        return p;
    };

    _Float16* wt_in = (_Float16*)take((size_t)D_MAIN * N_NUM * 2);
    _Float16* wt_e1 = (_Float16*)take((size_t)D_HID  * D_MAIN * 2);
    _Float16* wt_e2 = (_Float16*)take((size_t)D_MAIN * D_HID  * 2);
    _Float16* wt_s  = (_Float16*)take((size_t)D_MAIN * D_MAIN * 2);
    _Float16* wt_t1 = (_Float16*)take((size_t)D_HID  * D_MAIN * 2);
    _Float16* wt_t2 = (_Float16*)take((size_t)D_MAIN * D_HID  * 2);
    _Float16* wt_p1 = (_Float16*)take((size_t)D_HID  * D_MAIN * 2);
    _Float16* wt_p2 = (_Float16*)take((size_t)D_MAIN * D_HID  * 2);
    float*    x_enc = (float*)   take((size_t)NBATCH * D_MAIN * 4);
    float*    qk_f  = (float*)   take((size_t)NBATCH * D_MAIN * 4);
    _Float16* qk_h  = (_Float16*)take((size_t)NBATCH * D_MAIN * 2);
    _Float16* ck_h  = (_Float16*)take((size_t)N_CAND * D_MAIN * 2);
    float*    ck_n2 = (float*)   take((size_t)N_CAND * 4);
    float*    probs = (float*)   take((size_t)NBATCH * CTX * 4);
    int*      tidx  = (int*)     take((size_t)NBATCH * CTX * 4);
    float*    x_mix = (float*)   take((size_t)NBATCH * D_MAIN * 4);
    (void)ws_size; (void)n_in; (void)in_sizes; (void)out_size;

    // weight prep (f32 -> transposed f16)
    auto tr = [&](const float* W, _Float16* Wt, int K, int N) {
        int total = K * N;
        transpose_f16_kernel<<<(total + 255) / 256, 256, 0, stream>>>(W, Wt, K, N);
    };
    tr(W_in, wt_in, N_NUM, D_MAIN);
    tr(We1,  wt_e1, D_MAIN, D_HID);
    tr(We2,  wt_e2, D_HID,  D_MAIN);
    tr(Ws,   wt_s,  D_MAIN, D_MAIN);
    tr(Wt1,  wt_t1, D_MAIN, D_HID);
    tr(Wt2,  wt_t2, D_HID,  D_MAIN);
    tr(Wp1,  wt_p1, D_MAIN, D_HID);
    tr(Wp2,  wt_p2, D_HID,  D_MAIN);

    // encode queries (keep h, f32 keys, f16 keys)
    encode_kernel<true><<<NBATCH / 16, 256, 0, stream>>>(
        x_num, wt_in, b_in, wt_e1, be1, wt_e2, be2, g_m, b_m, wt_s, bs,
        x_enc, qk_f, qk_h, nullptr);

    // encode candidates (f16 keys + squared norms)
    encode_kernel<false><<<N_CAND / 16, 256, 0, stream>>>(
        cand_x, wt_in, b_in, wt_e1, be1, wt_e2, be2, g_m, b_m, wt_s, bs,
        nullptr, nullptr, ck_h, ck_n2);

    // streaming similarity + top-96 + softmax
    topk_kernel<<<NBATCH / 16, 256, 0, stream>>>(qk_h, ck_h, ck_n2, probs, tidx);

    // value MLP + weighted mix
    values_kernel<<<NBATCH, 256, 0, stream>>>(
        qk_f, ck_h, cand_y, tidx, probs, wt_t1, wt_t2, bt2, Wy, by, x_enc, x_mix);

    // predictor block + head
    head_kernel<<<NBATCH / 16, 256, 0, stream>>>(
        x_mix, wt_p1, bp1, wt_p2, bp2, g_h, b_h, Wh, bh, (float*)d_out);
}
